// EdgeAwareBiLSTM_4432406249925
// MI455X (gfx1250) — matrix operand order repaired and checked by reading, not compiled, not run on hardware
//
#include <hip/hip_runtime.h>
#include <hip/hip_bf16.h>

typedef __attribute__((ext_vector_type(16))) _Float16 v16h;
typedef __attribute__((ext_vector_type(8)))  _Float16 v8h;
typedef __attribute__((ext_vector_type(8)))  float    v8f;

#define B_  8
#define L_  16384
#define IN_ 64
#define H_  32

__device__ __forceinline__ float sigm(float x){ return 1.0f/(1.0f + __expf(-x)); }
__device__ __forceinline__ float tanh_f(float x){ return 2.0f*sigm(2.0f*x) - 1.0f; }

__device__ __forceinline__ v16h make_a(v8h lo, v8h hi){
  v16h a;
#pragma unroll
  for (int i=0;i<8;i++){ a[i]=lo[i]; a[8+i]=hi[i]; }
  return a;
}

// Load one A-matrix (16x32 f16) tile from a row-major [16][32] half matrix.
// ISA layout: lanes 0-15 -> row=lane, K halves {0..7, 16..23};
//             lanes 16-31 -> row=lane-16, K halves {8..15, 24..31}.
__device__ __forceinline__ v16h load_a_rowmajor(const _Float16* base, int lane){
  int row = lane & 15;
  int k0a = (lane & 16) ? 8 : 0;
  const _Float16* p = base + row*32;
  v8h lo = *(const v8h*)(p + k0a);
  v8h hi = *(const v8h*)(p + 16 + k0a);
  return make_a(lo, hi);
}

// Load a B-matrix column from 32 contiguous f32 weights (one K=32 row of W).
// Lane = column; lanes 0-15 hold K={0..7,16..23}, lanes 16-31 hold K={8..15,24..31}.
__device__ __forceinline__ v16h load_b_row32(const float* Wrow, int lane){
  int k0 = (lane & 16) ? 8 : 0;
  const float* p = Wrow;
  v16h b;
#pragma unroll
  for (int i=0;i<16;i++) b[i] = (_Float16)p[(i<8) ? (k0 + i) : (16 + k0 + (i-8))];
  return b;
}

// B-tile with contiguous columns: B[k][n] = W[nBase+n][kBase+k].
__device__ __forceinline__ v16h load_b_from_w(const float* W, int ldw, int nBase, int kBase, int lane){
  int n  = nBase + (lane & 15);
  return load_b_row32(W + n*ldw + kBase, lane);
}

// ---------------------------------------------------------------------------
// Kernel 1: projection  proj[t][b][h] = x_seq[b,t,:] @ proj_w.T + proj_b
// x is [B][IN][L]; output f16, padded to 16 batch rows per timestep, laid out
// exactly as the LSTM A-matrix wants it.  One wave = one timestep, 4 WMMAs.
// ---------------------------------------------------------------------------
__global__ __launch_bounds__(256) void proj_kernel(const float* __restrict__ x,
    const float* __restrict__ proj_w, const float* __restrict__ proj_b,
    _Float16* __restrict__ proj16)
{
  int lane = threadIdx.x & 31;
  int wave = threadIdx.x >> 5;
  int t = blockIdx.x * 8 + wave;

  int n_lo = lane & 15;
  v16h Bm[2][2];
#pragma unroll
  for (int tile=0; tile<2; ++tile)
#pragma unroll
    for (int ch=0; ch<2; ++ch)
      Bm[tile][ch] = load_b_from_w(proj_w, IN_, tile*16, ch*32, lane);

  v8f c0, c1;
  float bias0 = proj_b[n_lo], bias1 = proj_b[16 + n_lo];
#pragma unroll
  for (int r=0;r<8;r++){ c0[r]=bias0; c1[r]=bias1; }

  // Build A chunks directly from strided x (one-shot pass, L2-friendly along t).
  int brow = lane & 15;
  int k0a  = (lane & 16) ? 8 : 0;
  v16h a0, a1;
#pragma unroll
  for (int i=0;i<16;i++){
    int kc = (i<8) ? (k0a + i) : (16 + k0a + (i-8));
    float v0 = 0.0f, v1 = 0.0f;
    if (brow < B_) {
      size_t baseb = (size_t)brow * IN_ * L_ + (size_t)t;
      v0 = x[baseb + (size_t)kc      * L_];
      v1 = x[baseb + (size_t)(32+kc) * L_];
    }
    a0[i] = (_Float16)v0;
    a1[i] = (_Float16)v1;
  }

  v8f d0 = __builtin_amdgcn_wmma_f32_16x16x32_f16(false, a0, false, Bm[0][0], (short)0, c0, false, false);
  d0     = __builtin_amdgcn_wmma_f32_16x16x32_f16(false, a1, false, Bm[0][1], (short)0, d0, false, false);
  v8f d1 = __builtin_amdgcn_wmma_f32_16x16x32_f16(false, a0, false, Bm[1][0], (short)0, c1, false, false);
  d1     = __builtin_amdgcn_wmma_f32_16x16x32_f16(false, a1, false, Bm[1][1], (short)0, d1, false, false);

  int rbase = (lane & 16) ? 8 : 0;
  _Float16* outrow = proj16 + (size_t)t * 512;
#pragma unroll
  for (int r=0;r<8;r++){
    outrow[(rbase + r)*32 + n_lo]      = (_Float16)d0[r];
    outrow[(rbase + r)*32 + 16 + n_lo] = (_Float16)d1[r];
  }
}

// ---------------------------------------------------------------------------
// Kernel 2: BiLSTM recurrence.  One block per direction, 256 threads = 8 waves.
// B-columns are permuted so wave w computes ALL FOUR gates of hidden units
// 4w..4w+3:  column n <-> weight row (n>>2)*32 + 4w + (n&3).  Gate exchange is
// therefore wave-local (in-order LDS, no barrier); the h state is double-
// buffered in LDS so the whole step needs exactly ONE __syncthreads().
// ---------------------------------------------------------------------------
__global__ __launch_bounds__(256) void lstm_kernel(
    const _Float16* __restrict__ proj16,
    const float* __restrict__ wih_f, const float* __restrict__ whh_f,
    const float* __restrict__ bih_f, const float* __restrict__ bhh_f,
    const float* __restrict__ wih_b, const float* __restrict__ whh_b,
    const float* __restrict__ bih_b, const float* __restrict__ bhh_b,
    float* __restrict__ hout)
{
  const int dir = blockIdx.x;
  const float* wih = dir ? wih_b : wih_f;
  const float* whh = dir ? whh_b : whh_f;
  const float* bih = dir ? bih_b : bih_f;
  const float* bhh = dir ? bhh_b : bhh_f;

  int tid  = threadIdx.x;
  int lane = tid & 31;
  int wave = tid >> 5;
  int n_lo = lane & 15;

  // Column permutation: n -> (gate g = n>>2, hidden j = 4*wave + (n&3))
  int wrow = (n_lo >> 2)*32 + wave*4 + (n_lo & 3);   // row in [128][32] weights

  __shared__ __align__(16) _Float16 h16[2][16*32];   // double-buffered h (f16 A-tile)
  __shared__ float gatesLds[8][16*16];               // wave-private gate tiles

  for (int i = tid; i < 2*16*32; i += 256) (&h16[0][0])[i] = (_Float16)0.0f;

  // Persistent B-matrices (K=32 each) and bias accumulator.
  v16h b_x = load_b_row32(wih + wrow*32, lane);
  v16h b_h = load_b_row32(whh + wrow*32, lane);
  v8f c_bias;
  { float bv = bih[wrow] + bhh[wrow];
#pragma unroll
    for (int r=0;r<8;r++) c_bias[r]=bv; }

  // Update-phase assignment: one (batch, hidden) pair per lane.
  int b_upd = lane >> 2;                 // 0..7
  int jj    = lane & 3;
  int j_upd = wave*4 + jj;               // 0..31
  float cst = 0.0f;                      // f32 cell state in a register

  __syncthreads();

  int tt = dir ? (L_-1) : 0;
  v16h a_x = load_a_rowmajor(proj16 + (size_t)tt*512, lane);
  int rbase = (lane & 16) ? 8 : 0;
  float* gmine = &gatesLds[wave][b_upd*16 + jj];
  float* hbase = hout + dir*32 + j_upd;

  for (int step = 0; step < L_; ++step) {
    // --- matrix phase: gates = [proj_t | h] @ [wih ; whh] (per-wave N-tile)
    v16h a_h = load_a_rowmajor(h16[step & 1], lane);
    v8f d = __builtin_amdgcn_wmma_f32_16x16x32_f16(false, a_x, false, b_x, (short)0, c_bias, false, false);
    d     = __builtin_amdgcn_wmma_f32_16x16x32_f16(false, a_h, false, b_h, (short)0, d,      false, false);

    // software-pipeline next timestep's proj tile (independent of h chain)
    int stepn = step + 1;
    int ttn = dir ? (L_-1-stepn) : stepn;
    v16h a_xn = a_x;
    if (stepn < L_) a_xn = load_a_rowmajor(proj16 + (size_t)ttn*512, lane);

    // wave-private gate spill (in-order LDS, no cross-wave sync needed)
#pragma unroll
    for (int r=0;r<8;r++) gatesLds[wave][(rbase + r)*16 + n_lo] = d[r];

    // --- update phase: i,f,g,o for (b_upd, j_upd) all live in this wave
    float gi = gmine[0];
    float gf = gmine[4];
    float gg = gmine[8];
    float go = gmine[12];
    float ii = sigm(gi), ff = sigm(gf), gv = tanh_f(gg), oo = sigm(go);
    cst = ff*cst + ii*gv;
    float hv = oo * tanh_f(cst);
    h16[(step + 1) & 1][b_upd*32 + j_upd] = (_Float16)hv;
    hbase[((size_t)b_upd*L_ + (size_t)tt)*64] = hv;

    __syncthreads();          // publish new h before any wave reads it
    a_x = a_xn;
    tt  = ttn;
  }
}

// ---------------------------------------------------------------------------
// Kernel 3: boundary classifier  probs = sigmoid(w2 . relu(W1 h + b1) + b2)
// ---------------------------------------------------------------------------
__global__ __launch_bounds__(256) void boundary_kernel(
    const float* __restrict__ hout,
    const float* __restrict__ w1, const float* __restrict__ b1,
    const float* __restrict__ w2, const float* __restrict__ b2,
    float* __restrict__ probs)
{
  __shared__ float w1s[32*64];
  __shared__ float w2s[32];
  __shared__ float b1s[32];
  int tid = threadIdx.x;
  for (int i = tid; i < 2048; i += 256) w1s[i] = w1[i];
  if (tid < 32){ w2s[tid] = w2[tid]; b1s[tid] = b1[tid]; }
  __syncthreads();

  int row = blockIdx.x*256 + tid;      // row = b*L + t
  const float* hr = hout + (size_t)row*64;
  float xr[64];
#pragma unroll
  for (int k=0;k<16;k++) *(float4*)(xr + 4*k) = *(const float4*)(hr + 4*k);

  float acc = b2[0];
  for (int j2=0;j2<32;j2++){
    float s = b1s[j2];
    const float* wr = w1s + j2*64;
#pragma unroll
    for (int k=0;k<64;k++) s += xr[k]*wr[k];
    acc += w2s[j2]*fmaxf(s, 0.0f);
  }
  probs[row] = sigm(acc);
}

// ---------------------------------------------------------------------------
// Kernel 4: boundary-adjust stencil
// ---------------------------------------------------------------------------
__global__ __launch_bounds__(256) void adj_kernel(const float* __restrict__ probs,
                                                  float* __restrict__ adj)
{
  int row = blockIdx.x*256 + threadIdx.x;
  int t = row & (L_-1);
  float v = 0.0f;
  if (t > 0 && t < L_-1){
    float p  = probs[row];
    float lg = p - probs[row-1];
    float rg = p - probs[row+1];
    float alg = fabsf(lg), arg = fabsf(rg);
    if (lg < 0.0f && alg > arg)      v = -1.0f;
    else if (rg < 0.0f && arg > alg) v =  1.0f;
  }
  adj[row] = v;
}

// ---------------------------------------------------------------------------
// Kernel 5: mean-pool over L + real classifier.  One block per batch element.
// ---------------------------------------------------------------------------
__global__ __launch_bounds__(256) void real_kernel(
    const float* __restrict__ hout,
    const float* __restrict__ rc_w1, const float* __restrict__ rc_b1,
    const float* __restrict__ rc_w2, const float* __restrict__ rc_b2,
    float* __restrict__ realOut)
{
  int b = blockIdx.x;
  int tid = threadIdx.x;
  int j = tid & 63;
  int s = tid >> 6;                 // 4 stripes over time
  const float* base = hout + (size_t)b*L_*64 + j;
  float acc = 0.0f;
  for (int t = s; t < L_; t += 4) acc += base[(size_t)t*64];

  __shared__ float red[256];
  __shared__ float pooled[64];
  __shared__ float rh[32];
  red[tid] = acc;
  __syncthreads();
  if (tid < 64)
    pooled[tid] = (red[tid] + red[64+tid] + red[128+tid] + red[192+tid]) * (1.0f/L_);
  __syncthreads();
  if (tid < 32){
    float sacc = rc_b1[tid];
#pragma unroll
    for (int k=0;k<64;k++) sacc += pooled[k]*rc_w1[tid*64+k];
    rh[tid] = fmaxf(sacc, 0.0f);
  }
  __syncthreads();
  if (tid == 0){
    float a2 = rc_b2[0];
#pragma unroll
    for (int k=0;k<32;k++) a2 += rh[k]*rc_w2[k];
    realOut[b] = sigm(a2);
  }
}

// ---------------------------------------------------------------------------
extern "C" void kernel_launch(void* const* d_in, const int* in_sizes, int n_in,
                              void* d_out, int out_size, void* d_ws, size_t ws_size,
                              hipStream_t stream)
{
  (void)in_sizes; (void)n_in; (void)out_size; (void)ws_size;
  const float* x      = (const float*)d_in[0];
  const float* proj_w = (const float*)d_in[1];
  const float* proj_b = (const float*)d_in[2];
  const float* wih_f  = (const float*)d_in[3];
  const float* whh_f  = (const float*)d_in[4];
  const float* bih_f  = (const float*)d_in[5];
  const float* bhh_f  = (const float*)d_in[6];
  const float* wih_b  = (const float*)d_in[7];
  const float* whh_b  = (const float*)d_in[8];
  const float* bih_b  = (const float*)d_in[9];
  const float* bhh_b  = (const float*)d_in[10];
  const float* bc_w1  = (const float*)d_in[11];
  const float* bc_b1  = (const float*)d_in[12];
  const float* bc_w2  = (const float*)d_in[13];
  const float* bc_b2  = (const float*)d_in[14];
  const float* rc_w1  = (const float*)d_in[15];
  const float* rc_b1  = (const float*)d_in[16];
  const float* rc_w2  = (const float*)d_in[17];
  const float* rc_b2  = (const float*)d_in[18];

  _Float16* proj16 = (_Float16*)d_ws;                           // 16 MB
  float*    hout   = (float*)((char*)d_ws + (size_t)(16u<<20)); // 32 MB

  float* probs = (float*)d_out;
  float* adj   = probs + B_*L_;
  float* realO = adj   + B_*L_;

  proj_kernel<<<L_/8, 256, 0, stream>>>(x, proj_w, proj_b, proj16);
  lstm_kernel<<<2, 256, 0, stream>>>(proj16, wih_f, whh_f, bih_f, bhh_f,
                                     wih_b, whh_b, bih_b, bhh_b, hout);
  boundary_kernel<<<(B_*L_)/256, 256, 0, stream>>>(hout, bc_w1, bc_b1, bc_w2, bc_b2, probs);
  adj_kernel<<<(B_*L_)/256, 256, 0, stream>>>(probs, adj);
  real_kernel<<<B_, 256, 0, stream>>>(hout, rc_w1, rc_b1, rc_w2, rc_b2, realO);
}
